// Block_30657476559040
// MI455X (gfx1250) — compile-verified, matrix operand order checked by reading, not run on hardware
//
#include <hip/hip_runtime.h>

// ---------------------------------------------------------------------------
// CDNA5 WMMA types (wave32): v_wmma_f32_16x16x32_bf16
// ---------------------------------------------------------------------------
typedef __attribute__((ext_vector_type(16))) __bf16 v16bf;
typedef __attribute__((ext_vector_type(8)))  float  v8f;

struct U8 { unsigned int u[8]; };

static __device__ inline v16bf frag_cast(const U8& f) {
    return __builtin_bit_cast(v16bf, f);
}

static __device__ inline unsigned short f2bf(float f) {
    unsigned int u = __builtin_bit_cast(unsigned int, f);
    u += 0x7FFFu + ((u >> 16) & 1u);      // round-to-nearest-even
    return (unsigned short)(u >> 16);
}

static __device__ inline float wave_sum(float v) {
    #pragma unroll
    for (int m = 16; m >= 1; m >>= 1) v += __shfl_xor(v, m, 32);
    return v;
}

// K offset inside a 16x32 bf16 A/B fragment for dword v, lane-half h (ISA 7.12.2)
static __device__ inline int frag_k(int v, int half) {
    return ((v < 4) ? 2 * v : 16 + 2 * (v - 4)) + 8 * half;
}

// Load an A-fragment: two 16B-contiguous runs per lane.
static __device__ inline U8 load_a_frag(const unsigned short* row, int k0, int half) {
    uint4 a0 = *(const uint4*)(row + k0 + 8 * half);
    uint4 a1 = *(const uint4*)(row + k0 + 16 + 8 * half);
    U8 f;
    f.u[0] = a0.x; f.u[1] = a0.y; f.u[2] = a0.z; f.u[3] = a0.w;
    f.u[4] = a1.x; f.u[5] = a1.y; f.u[6] = a1.z; f.u[7] = a1.w;
    return f;
}

// Load a swizzled B-fragment (32 contiguous bytes per lane) from any memory.
static __device__ inline U8 load_b_frag(const unsigned int* sw, size_t tileIdx, int lane) {
    const uint4* p = (const uint4*)(sw + (tileIdx * 32 + lane) * 8);
    uint4 b0 = p[0], b1 = p[1];
    U8 f;
    f.u[0] = b0.x; f.u[1] = b0.y; f.u[2] = b0.z; f.u[3] = b0.w;
    f.u[4] = b1.x; f.u[5] = b1.y; f.u[6] = b1.z; f.u[7] = b1.w;
    return f;
}

// ---------------------------------------------------------------------------
// CDNA5 async global->LDS staging (ASYNCcnt path)
// ---------------------------------------------------------------------------
static __device__ inline void async_b128_to_lds(unsigned int lds_byte_addr,
                                                const void* gaddr) {
    asm volatile("global_load_async_to_lds_b128 %0, %1, off"
                 :: "v"(lds_byte_addr), "v"((unsigned long long)(uintptr_t)gaddr)
                 : "memory");
}
static __device__ inline void wait_async0() {
    asm volatile("s_wait_asynccnt 0" ::: "memory");
}

// Stage one k-step of swizzled B (4 ntiles x 32 lanes x 32B = 4KB) into LDS:
// exactly one b128 chunk per thread of a 256-thread block.
static __device__ inline void stage_b(const unsigned int* __restrict__ Bsw,
                                      unsigned int* smbase, int kt, int ntN,
                                      int ns, int tid) {
    int nt4  = tid >> 6;          // 0..3
    int r    = tid & 63;
    int lane = r >> 1;            // 0..31
    int h16  = r & 1;             // which 16B half of the 32B fragment
    const unsigned int* gsrc =
        Bsw + (((size_t)kt * ntN + ns * 4 + nt4) * 32 + lane) * 8 + h16 * 4;
    unsigned int lds = (unsigned int)(uintptr_t)(smbase + tid * 4); // tid*16 bytes
    async_b128_to_lds(lds, gsrc);
}

#define DM   384
#define DFF  1536
#define TLEN 256
#define NH   6
#define HS   64
#define MROWS (128 * 256)   // B*T = 32768

// ---------------------------------------------------------------------------
// Weight prep
// ---------------------------------------------------------------------------
__global__ void k_pack_qkv(const float* __restrict__ Wq,
                           const float* __restrict__ Wk,
                           const float* __restrict__ Wv,
                           unsigned short* __restrict__ wqkv) {
    int idx = blockIdx.x * 256 + threadIdx.x;         // over 384*1152
    if (idx >= DM * 3 * DM) return;
    int d = idx / (3 * DM);
    int n = idx - d * (3 * DM);
    int part = n / DM;                                 // 0=q 1=k 2=v
    int nn = n - part * DM;
    int h = nn >> 6, e = nn & 63;
    const float* W = (part == 0) ? Wq : (part == 1) ? Wk : Wv;
    wqkv[idx] = f2bf(W[((size_t)h * DM + d) * HS + e]);   // W[h][d][e]
}

__global__ void k_cvt_bf16(const float* __restrict__ src,
                           unsigned short* __restrict__ dst, int n) {
    int i = blockIdx.x * 256 + threadIdx.x;
    if (i < n) dst[i] = f2bf(src[i]);
}

// bf16 row-major [K,N] -> fragment-swizzled [kt][nt][lane][8 dwords]
__global__ void k_swizzle_b(const unsigned short* __restrict__ src,
                            unsigned int* __restrict__ dst, int K, int N) {
    int idx = blockIdx.x * 256 + threadIdx.x;
    int total = (K >> 5) * (N >> 4) * 256;
    if (idx >= total) return;
    int vv   = idx & 7;
    int lane = (idx >> 3) & 31;
    int tile = idx >> 8;
    int ntN  = N >> 4;
    int kt = tile / ntN, nt = tile - kt * ntN;
    int kk = kt * 32 + frag_k(vv, lane >> 4);
    int n  = nt * 16 + (lane & 15);
    dst[idx] = (unsigned int)src[(size_t)kk * N + n]
             | ((unsigned int)src[(size_t)(kk + 1) * N + n] << 16);
}

// ---------------------------------------------------------------------------
// LayerNorm: one wave per 384-wide row, fp32 in -> bf16 out
// ---------------------------------------------------------------------------
__global__ void k_layernorm(const float* __restrict__ x,
                            const float* __restrict__ g,
                            const float* __restrict__ b,
                            unsigned short* __restrict__ out, int nrows) {
    int lane = threadIdx.x & 31, wave = threadIdx.x >> 5;
    int row = blockIdx.x * (blockDim.x >> 5) + wave;
    if (row >= nrows) return;
    const float* xr = x + (size_t)row * DM;
    float vals[12];
    float s = 0.f;
    #pragma unroll
    for (int i = 0; i < 12; ++i) { vals[i] = xr[lane + 32 * i]; s += vals[i]; }
    float mu = wave_sum(s) * (1.0f / DM);
    float vr = 0.f;
    #pragma unroll
    for (int i = 0; i < 12; ++i) { float d = vals[i] - mu; vr += d * d; }
    float rstd = __frsqrt_rn(wave_sum(vr) * (1.0f / DM) + 1e-5f);
    unsigned short* orow = out + (size_t)row * DM;
    #pragma unroll
    for (int i = 0; i < 12; ++i) {
        int c = lane + 32 * i;
        orow[c] = f2bf((vals[i] - mu) * rstd * g[c] + b[c]);
    }
}

// ---------------------------------------------------------------------------
// WMMA GEMM: block = 8 waves = 8 M-tiles sharing one 64-wide N strip.
// B fragments async-staged global->LDS, double-buffered; A via b128 loads.
// ---------------------------------------------------------------------------
__global__ __launch_bounds__(256)
void k_gemm(const unsigned short* __restrict__ A,
            const unsigned int* __restrict__ Bsw,
            int M, int N, int K,
            const float* __restrict__ bias,
            const float* __restrict__ resid,
            float* __restrict__ outf,
            unsigned short* __restrict__ outb,
            int relu) {
    __shared__ __align__(16) unsigned int bsh[2][1024];   // 2 x 4KB
    int tid = threadIdx.x;
    int lane = tid & 31, wave = tid >> 5;
    int nstr = N >> 6;
    int bm = blockIdx.x / nstr, ns = blockIdx.x - bm * nstr;
    int mt = bm * 8 + wave;
    int half = lane >> 4, l15 = lane & 15;
    int ntN = N >> 4;

    const unsigned short* arow = A + (size_t)(mt * 16 + l15) * K;
    v8f acc[4];
    v8f z = {};
    #pragma unroll
    for (int i = 0; i < 4; ++i) acc[i] = z;

    int nkt = K >> 5;
    stage_b(Bsw, bsh[0], 0, ntN, ns, tid);
    for (int kt = 0; kt < nkt; ++kt) {
        wait_async0();
        __syncthreads();                     // staged buffer visible to block
        if (kt + 1 < nkt) stage_b(Bsw, bsh[(kt + 1) & 1], kt + 1, ntN, ns, tid);
        int k0 = kt * 32;
        if (k0 + 32 < K) __builtin_prefetch(arow + k0 + 32, 0, 0);
        U8 au = load_a_frag(arow, k0, half);
        v16bf av = frag_cast(au);
        const unsigned int* smb = bsh[kt & 1];
        #pragma unroll
        for (int nt4 = 0; nt4 < 4; ++nt4) {
            U8 bu = load_b_frag(smb, nt4, lane);
            acc[nt4] = __builtin_amdgcn_wmma_f32_16x16x32_bf16(
                false, av, false, frag_cast(bu), (short)0, acc[nt4], false, false);
        }
    }
    #pragma unroll
    for (int nt4 = 0; nt4 < 4; ++nt4)
        #pragma unroll
        for (int r = 0; r < 8; ++r) {
            int m = mt * 16 + r + 8 * half;
            int n = ns * 64 + nt4 * 16 + l15;
            float v = acc[nt4][r];
            if (bias)  v += bias[n];
            if (resid) v += resid[(size_t)m * N + n];
            if (relu)  v = fmaxf(v, 0.0f);
            if (outf) outf[(size_t)m * N + n] = v;
            else      outb[(size_t)m * N + n] = f2bf(v);
        }
}

// ---------------------------------------------------------------------------
// Fused QKV GEMM (same async-staged scheme): q,k row-major [B,H,T,E] bf16,
// V scattered directly into PV-fragment layout Vsw[bh][kt(8)][nt(4)][lane][8].
// ---------------------------------------------------------------------------
__global__ __launch_bounds__(256)
void k_gemm_qkv(const unsigned short* __restrict__ A,
                const unsigned int* __restrict__ Bsw,
                unsigned short* __restrict__ q,
                unsigned short* __restrict__ k,
                unsigned int* __restrict__ vsw) {
    const int N = 3 * DM, K = DM, nstr = N >> 6, ntN = N >> 4;
    __shared__ __align__(16) unsigned int bsh[2][1024];
    int tid = threadIdx.x;
    int lane = tid & 31, wave = tid >> 5;
    int bm = blockIdx.x / nstr, ns = blockIdx.x - bm * nstr;
    int mt = bm * 8 + wave;
    int half = lane >> 4, l15 = lane & 15;

    const unsigned short* arow = A + (size_t)(mt * 16 + l15) * K;
    v8f acc[4];
    v8f z = {};
    #pragma unroll
    for (int i = 0; i < 4; ++i) acc[i] = z;

    int nkt = K >> 5;
    stage_b(Bsw, bsh[0], 0, ntN, ns, tid);
    for (int kt = 0; kt < nkt; ++kt) {
        wait_async0();
        __syncthreads();
        if (kt + 1 < nkt) stage_b(Bsw, bsh[(kt + 1) & 1], kt + 1, ntN, ns, tid);
        int k0 = kt * 32;
        if (k0 + 32 < K) __builtin_prefetch(arow + k0 + 32, 0, 0);
        U8 au = load_a_frag(arow, k0, half);
        v16bf av = frag_cast(au);
        const unsigned int* smb = bsh[kt & 1];
        #pragma unroll
        for (int nt4 = 0; nt4 < 4; ++nt4) {
            U8 bu = load_b_frag(smb, nt4, lane);
            acc[nt4] = __builtin_amdgcn_wmma_f32_16x16x32_bf16(
                false, av, false, frag_cast(bu), (short)0, acc[nt4], false, false);
        }
    }

    int bb = (mt * 16) >> 8;                    // batch index (tile never crosses)
    #pragma unroll
    for (int nt4 = 0; nt4 < 4; ++nt4) {
        int n0 = ns * 64 + nt4 * 16;            // 0..1151, multiple of 16
        int part = n0 / DM;
        int nn0 = n0 - part * DM;               // 0..383
        int h = nn0 >> 6;
        if (part < 2) {
            unsigned short* dst = (part == 0) ? q : k;
            #pragma unroll
            for (int r = 0; r < 8; ++r) {
                int m = mt * 16 + r + 8 * half;
                int t = m & 255;
                int e = (nn0 & 63) + l15;
                dst[(((size_t)bb * NH + h) * TLEN + t) * HS + e] = f2bf(acc[nt4][r]);
            }
        } else {
            int e = (nn0 & 63) + l15;           // 0..63
            int nt = e >> 4;
            int bh = bb * NH + h;
            #pragma unroll
            for (int j = 0; j < 4; ++j) {
                int t = (mt * 16 + 8 * half + 2 * j) & 255;   // even row of pair
                int kt2 = t >> 5;
                int r5 = t & 31;
                int half2 = (r5 >> 3) & 1;
                int rr = r5 - 8 * half2;        // {0,2,4,6,16,18,20,22}
                int vv = (rr < 8) ? (rr >> 1) : 4 + ((rr - 16) >> 1);
                int lane2 = half2 * 16 + (e & 15);
                unsigned int pk = (unsigned int)f2bf(acc[nt4][2 * j])
                                | ((unsigned int)f2bf(acc[nt4][2 * j + 1]) << 16);
                vsw[((((size_t)bh * 8 + kt2) * 4 + nt) * 32 + lane2) * 8 + vv] = pk;
            }
        }
    }
}

// ---------------------------------------------------------------------------
// Causal attention: one wave per (b,h, 16-row) tile.
// ---------------------------------------------------------------------------
#define AWAVES 2
__global__ __launch_bounds__(32 * AWAVES)
void k_attn(const unsigned short* __restrict__ q,
            const unsigned short* __restrict__ k,
            const unsigned int* __restrict__ vsw,
            unsigned short* __restrict__ attn) {
    __shared__ float sc[AWAVES][16][TLEN];
    int lane = threadIdx.x & 31, wave = threadIdx.x >> 5;
    int g  = blockIdx.x * AWAVES + wave;
    int bh = g >> 4;                  // b*NH + h
    int rt = g & 15;
    int r0 = rt << 4;
    int half = lane >> 4, l15 = lane & 15;
    const unsigned short* qb = q + (size_t)bh * TLEN * HS;
    const unsigned short* kb = k + (size_t)bh * TLEN * HS;

    float* scw = &sc[wave][0][0];
    for (int i = lane; i < 16 * TLEN; i += 32) scw[i] = 0.f;
    __syncthreads();

    const unsigned short* qrow = qb + (size_t)(r0 + l15) * HS;
    U8 aq[2];
    #pragma unroll
    for (int ks = 0; ks < 2; ++ks) aq[ks] = load_a_frag(qrow, ks * 32, half);

    const int limit = r0 + 16;
    for (int s0 = 0; s0 < limit; s0 += 16) {
        const unsigned short* krow = kb + (size_t)(s0 + l15) * HS;
        v8f c = {};
        #pragma unroll
        for (int ks = 0; ks < 2; ++ks) {
            U8 bu = load_a_frag(krow, ks * 32, half);
            c = __builtin_amdgcn_wmma_f32_16x16x32_bf16(
                false, frag_cast(aq[ks]), false, frag_cast(bu), (short)0, c, false, false);
        }
        #pragma unroll
        for (int r = 0; r < 8; ++r) {
            int ml = r + 8 * half;
            float val = c[r] * 0.125f;
            if (s0 + l15 > r0 + ml) val = -3.0e38f;
            sc[wave][ml][s0 + l15] = val;
        }
    }
    __syncthreads();

    {
        int row = l15;
        float mx = -3.0e38f;
        for (int cc = half; cc < limit; cc += 2) mx = fmaxf(mx, sc[wave][row][cc]);
        mx = fmaxf(mx, __shfl_xor(mx, 16, 32));
        float sm = 0.f;
        for (int cc = half; cc < limit; cc += 2) {
            float e = __expf(sc[wave][row][cc] - mx);
            sc[wave][row][cc] = e;
            sm += e;
        }
        sm += __shfl_xor(sm, 16, 32);
        float inv = 1.0f / sm;
        for (int cc = half; cc < limit; cc += 2) sc[wave][row][cc] *= inv;
    }
    __syncthreads();

    v8f o[4];
    v8f z = {};
    #pragma unroll
    for (int i = 0; i < 4; ++i) o[i] = z;
    int ksteps = (limit + 31) >> 5;
    for (int ks = 0; ks < ksteps; ++ks) {
        U8 au;
        #pragma unroll
        for (int vv = 0; vv < 8; ++vv) {
            int kk = ks * 32 + frag_k(vv, half);
            au.u[vv] = (unsigned int)f2bf(sc[wave][l15][kk])
                     | ((unsigned int)f2bf(sc[wave][l15][kk + 1]) << 16);
        }
        v16bf av = frag_cast(au);
        #pragma unroll
        for (int nt = 0; nt < 4; ++nt) {
            U8 bu = load_b_frag(vsw, ((size_t)bh * 8 + ks) * 4 + nt, lane);
            o[nt] = __builtin_amdgcn_wmma_f32_16x16x32_bf16(
                false, av, false, frag_cast(bu), (short)0, o[nt], false, false);
        }
    }

    int bb = bh / NH, hh = bh - bb * NH;
    #pragma unroll
    for (int nt = 0; nt < 4; ++nt)
        #pragma unroll
        for (int r = 0; r < 8; ++r) {
            int t = r0 + r + 8 * half;
            int e = hh * HS + nt * 16 + l15;
            attn[((size_t)bb * TLEN + t) * DM + e] = f2bf(o[nt][r]);
        }
}

// ---------------------------------------------------------------------------
// Host-side orchestration
// ---------------------------------------------------------------------------
#define MB ((size_t)1 << 20)

extern "C" void kernel_launch(void* const* d_in, const int* in_sizes, int n_in,
                              void* d_out, int out_size, void* d_ws, size_t ws_size,
                              hipStream_t stream) {
    const float* x   = (const float*)d_in[0];
    const float* Wq  = (const float*)d_in[1];
    const float* Wk  = (const float*)d_in[2];
    const float* Wv  = (const float*)d_in[3];
    const float* Wp  = (const float*)d_in[4];
    const float* bp  = (const float*)d_in[5];
    const float* W1  = (const float*)d_in[6];
    const float* b1  = (const float*)d_in[7];
    const float* W2  = (const float*)d_in[8];
    const float* b2  = (const float*)d_in[9];
    const float* g1  = (const float*)d_in[10];
    const float* be1 = (const float*)d_in[11];
    const float* g2  = (const float*)d_in[12];
    const float* be2 = (const float*)d_in[13];
    (void)in_sizes; (void)n_in; (void)out_size; (void)ws_size;

    char* ws = (char*)d_ws;
    unsigned short* HBF  = (unsigned short*)(ws + 0 * MB);   // dead after QKV
    unsigned short* Q    = (unsigned short*)(ws + 32 * MB);  // dead after attn
    unsigned short* Kb   = (unsigned short*)(ws + 64 * MB);
    unsigned int*   Vsw  = (unsigned int*)  (ws + 96 * MB);  // frag-swizzled V
    unsigned short* FF1  = (unsigned short*)(ws + 0 * MB);   // reuses HBF/Q/K
    unsigned short* ATTN = (unsigned short*)(ws + 128 * MB);
    unsigned short* H2   = (unsigned short*)(ws + 156 * MB);
    float*          X1   = (float*)        (ws + 184 * MB);  // live to the end
    unsigned int*   WQKVs= (unsigned int*)  (ws + 236 * MB);
    unsigned int*   WPs  = (unsigned int*)  (ws + 238 * MB);
    unsigned int*   W1s  = (unsigned int*)  (ws + 239 * MB);
    unsigned int*   W2s  = (unsigned int*)  (ws + 241 * MB);
    unsigned short* TMP  = (unsigned short*)(ws + 244 * MB); // weight staging

    // 1) weight prep: convert -> row-major bf16 (TMP) -> fragment swizzle
    k_pack_qkv<<<(DM * 3 * DM + 255) / 256, 256, 0, stream>>>(Wq, Wk, Wv, TMP);
    k_swizzle_b<<<(DM * 3 * DM / 2 + 255) / 256, 256, 0, stream>>>(TMP, WQKVs, DM, 3 * DM);
    k_cvt_bf16<<<(DM * DM + 255) / 256, 256, 0, stream>>>(Wp, TMP, DM * DM);
    k_swizzle_b<<<(DM * DM / 2 + 255) / 256, 256, 0, stream>>>(TMP, WPs, DM, DM);
    k_cvt_bf16<<<(DM * DFF + 255) / 256, 256, 0, stream>>>(W1, TMP, DM * DFF);
    k_swizzle_b<<<(DM * DFF / 2 + 255) / 256, 256, 0, stream>>>(TMP, W1s, DM, DFF);
    k_cvt_bf16<<<(DFF * DM + 255) / 256, 256, 0, stream>>>(W2, TMP, DFF * DM);
    k_swizzle_b<<<(DFF * DM / 2 + 255) / 256, 256, 0, stream>>>(TMP, W2s, DFF, DM);

    // 2) LN1 -> bf16
    k_layernorm<<<MROWS / 8, 256, 0, stream>>>(x, g1, be1, HBF, MROWS);

    // 3) fused QKV projection (V lands pre-swizzled for PV)
    k_gemm_qkv<<<(MROWS / 128) * (3 * DM / 64), 256, 0, stream>>>(HBF, WQKVs, Q, Kb, Vsw);

    // 4) causal attention
    {
        int rowTiles = 128 * NH * (TLEN / 16);          // 12288
        k_attn<<<rowTiles / AWAVES, 32 * AWAVES, 0, stream>>>(Q, Kb, Vsw, ATTN);
    }

    // 5) output projection + residual -> X1 (fp32)
    k_gemm<<<(MROWS / 128) * (DM / 64), 256, 0, stream>>>(ATTN, WPs, MROWS, DM, DM,
                                                          bp, x, X1, nullptr, 0);

    // 6) LN2 -> bf16
    k_layernorm<<<MROWS / 8, 256, 0, stream>>>(X1, g2, be2, H2, MROWS);

    // 7) FFN1 (+bias, ReLU) -> bf16
    k_gemm<<<(MROWS / 128) * (DFF / 64), 256, 0, stream>>>(H2, W1s, MROWS, DFF, DM,
                                                           b1, nullptr, nullptr, FF1, 1);

    // 8) FFN2 (+bias, +residual X1) -> d_out fp32
    k_gemm<<<(MROWS / 128) * (DM / 64), 256, 0, stream>>>(FF1, W2s, MROWS, DM, DFF,
                                                          b2, X1, (float*)d_out, nullptr, 0);
}
